// SSD_39058432590092
// MI455X (gfx1250) — compile-verified, matrix operand order checked by reading, not run on hardware
//
#include <hip/hip_runtime.h>
#include <stdint.h>

#define NUM_PRIORS_C 72192
#define BATCH_C      64
#define TOPK_C       200
#define KPAD_C       208      // 13 * 16, WMMA tile padding
#define NTILE_C      13
#define CONF_T       0.01f
#define NMS_T        0.45f
#define CAND_CAP     512

typedef __attribute__((ext_vector_type(2)))  float    v2f;
typedef __attribute__((ext_vector_type(8)))  float    v8f;
typedef __attribute__((ext_vector_type(16))) _Float16 v16h;

#if defined(__has_builtin)
#if __has_builtin(__builtin_amdgcn_wmma_f32_16x16x4_f32)
#define HAVE_WMMA_F32X4 1
#endif
#endif

// ---------------------------------------------------------------------------
// Zero the whole output (class-0 slice stays zero; dropped rows stay zero).
// ---------------------------------------------------------------------------
__global__ __launch_bounds__(256) void zero_f32(float* __restrict__ o, int n) {
  int i = blockIdx.x * 256 + threadIdx.x;
  if (i < n) o[i] = 0.0f;
}

// ---------------------------------------------------------------------------
// Kernel 1: per-batch exact top-200 via byte-radix select + bitonic sort.
// One workgroup (8 wave32s) per batch; conf is L2-resident across passes.
// ---------------------------------------------------------------------------
__global__ __launch_bounds__(256) void topk_select(const float* __restrict__ conf,
                                                   int* __restrict__ wsI,
                                                   float* __restrict__ wsS,
                                                   int P) {
  __shared__ unsigned hist[256];
  __shared__ unsigned long long cand[CAND_CAP];
  __shared__ unsigned s_prefix;
  __shared__ int s_r, s_total, s_cnt;

  const int b = blockIdx.x;
  const int tid = threadIdx.x;
  const float* __restrict__ cp = conf + (size_t)b * (size_t)P * 2;  // score = cp[2p+1]

  if (tid == 0) { s_prefix = 0u; s_r = TOPK_C; s_total = 0; }

  // 4 radix passes (bits 31..24 down to 7..0) to find the exact 200th key.
  for (int pass = 0; pass < 4; ++pass) {
    __syncthreads();
    if (pass > 0 && s_total < TOPK_C) break;   // fewer than 200 valid: accept all
    const unsigned prefix = s_prefix;
    const int shift = 24 - pass * 8;
    hist[tid] = 0u;
    __syncthreads();
    for (int p = tid; p < P; p += 256) {
      if (pass == 0 && p + 8192 < P)
        __builtin_prefetch((const void*)&cp[2 * (p + 8192) + 1], 0, 0);
      float s = cp[2 * p + 1];
      if (s > CONF_T) {
        unsigned k = __float_as_uint(s);  // positive floats: monotone as uint
        bool match = (pass == 0) || ((k >> (shift + 8)) == (prefix >> (shift + 8)));
        if (match) atomicAdd(&hist[(k >> shift) & 0xFFu], 1u);
      }
    }
    __syncthreads();
    if (tid == 0) {
      if (pass == 0) {
        int tot = 0;
        for (int i = 0; i < 256; ++i) tot += (int)hist[i];
        s_total = tot;
      }
      if (s_total >= TOPK_C) {
        int r = s_r, cum = 0, bin = 255;
        for (; bin >= 0; --bin) { cum += (int)hist[bin]; if (cum >= r) break; }
        if (bin < 0) bin = 0;
        s_prefix = prefix | ((unsigned)bin << shift);
        s_r = r - (cum - (int)hist[bin]);
      }
    }
  }
  __syncthreads();
  if (tid == 0) s_cnt = 0;
  __syncthreads();

  // Collect all candidates with key >= threshold.
  const unsigned thr = s_prefix;
  for (int p = tid; p < P; p += 256) {
    float s = cp[2 * p + 1];
    if (s > CONF_T) {
      unsigned k = __float_as_uint(s);
      if (k >= thr) {
        int pos = atomicAdd(&s_cnt, 1);
        if (pos < CAND_CAP)
          cand[pos] = (((unsigned long long)(~k)) << 32) | (unsigned)p;
      }
    }
  }
  __syncthreads();
  int cnt = s_cnt; if (cnt > CAND_CAP) cnt = CAND_CAP;
  for (int i = tid; i < CAND_CAP; i += 256)
    if (i >= cnt) cand[i] = ~0ULL;
  __syncthreads();

  // Bitonic sort ascending on composite key => score desc, index asc (jax tie order).
  for (int k2 = 2; k2 <= CAND_CAP; k2 <<= 1) {
    for (int jst = k2 >> 1; jst > 0; jst >>= 1) {
      for (int i = tid; i < CAND_CAP; i += 256) {
        int l = i ^ jst;
        if (l > i) {
          bool up = ((i & k2) == 0);
          unsigned long long a = cand[i], c = cand[l];
          if ((a > c) == up) { cand[i] = c; cand[l] = a; }
        }
      }
      __syncthreads();
    }
  }

  const int keff = (s_total < TOPK_C) ? s_total : TOPK_C;
  if (tid < KPAD_C) {
    if (tid < keff) {
      unsigned long long v = cand[tid];
      wsI[b * KPAD_C + tid] = (int)(unsigned)(v & 0xFFFFFFFFull);
      wsS[b * KPAD_C + tid] = __uint_as_float(~(unsigned)(v >> 32));
    } else {
      wsI[b * KPAD_C + tid] = 0;
      wsS[b * KPAD_C + tid] = -1.0f;   // invalid sentinel (keep=false)
    }
  }
}

// ---------------------------------------------------------------------------
// Kernel 2: gather+decode the 200 winners (async global->LDS), WMMA outer-sum
// of areas for the union term, IoU suppression bitmask, greedy NMS, compact.
// ---------------------------------------------------------------------------
__global__ __launch_bounds__(256) void decode_nms(const float* __restrict__ loc,
                                                  const float* __restrict__ prior,
                                                  const int* __restrict__ wsI,
                                                  const float* __restrict__ wsS,
                                                  float* __restrict__ out,
                                                  int P) {
  __shared__ float s_score[KPAD_C];
  __shared__ alignas(16) float s_locraw[TOPK_C * 4];
  __shared__ alignas(16) float s_priraw[TOPK_C * 4];
  __shared__ float s_box[KPAD_C * 4];
  __shared__ float s_area[KPAD_C];
  __shared__ unsigned s_sup[KPAD_C * 8];  // 208 rows x 256-bit suppression mask
  __shared__ unsigned s_keep[8];
  __shared__ int s_pos[KPAD_C];

  const int b = blockIdx.x;
  const int tid = threadIdx.x;

  for (int i = tid; i < KPAD_C * 8; i += 256) s_sup[i] = 0u;
  if (tid < 8) s_keep[tid] = 0u;

  const int j = tid;
  float sc = -1.0f;
  int idx = 0;
  if (j < KPAD_C) {
    sc = wsS[b * KPAD_C + j];
    idx = wsI[b * KPAD_C + j];
    s_score[j] = sc;
  }
  const bool valid = (j < TOPK_C) && (sc > CONF_T);

  // Async gather: 16B of loc + 16B of prior per candidate, straight into LDS.
  if (valid) {
    unsigned loff = (unsigned)(((size_t)b * (size_t)P + (size_t)idx) * 16u);
    unsigned poff = (unsigned)((size_t)idx * 16u);
    unsigned lds_l = (unsigned)(size_t)(&s_locraw[j * 4]);  // low 32b = LDS offset
    unsigned lds_p = (unsigned)(size_t)(&s_priraw[j * 4]);
    unsigned long long lbase = (unsigned long long)(size_t)loc;
    unsigned long long pbase = (unsigned long long)(size_t)prior;
    asm volatile("global_load_async_to_lds_b128 %0, %1, %2 offset:0"
                 :: "v"(lds_l), "v"(loff), "s"(lbase) : "memory");
    asm volatile("global_load_async_to_lds_b128 %0, %1, %2 offset:0"
                 :: "v"(lds_p), "v"(poff), "s"(pbase) : "memory");
  }
  asm volatile("s_wait_asynccnt 0x0" ::: "memory");
  __syncthreads();

  // Decode (cx,cy,w,h)->corners; padded/invalid rows become degenerate boxes.
  if (j < KPAD_C) {
    float x1 = 0.f, y1 = 0.f, x2 = 0.f, y2 = 0.f;
    if (valid) {
      float lx = s_locraw[j*4+0], ly = s_locraw[j*4+1];
      float lw = s_locraw[j*4+2], lh = s_locraw[j*4+3];
      float pcx = s_priraw[j*4+0], pcy = s_priraw[j*4+1];
      float pw  = s_priraw[j*4+2], ph  = s_priraw[j*4+3];
      float cx = pcx + lx * 0.1f * pw;
      float cy = pcy + ly * 0.1f * ph;
      float w  = pw * expf(lw * 0.2f);
      float h  = ph * expf(lh * 0.2f);
      x1 = cx - 0.5f * w; y1 = cy - 0.5f * h;
      x2 = cx + 0.5f * w; y2 = cy + 0.5f * h;
      atomicOr(&s_keep[j >> 5], 1u << (j & 31));
    }
    s_box[j*4+0] = x1; s_box[j*4+1] = y1; s_box[j*4+2] = x2; s_box[j*4+3] = y2;
    s_area[j] = (x2 - x1) * (y2 - y1);
  }
  __syncthreads();

  // WMMA: U[i][j] = area_i + area_j as a rank-2 16x16x4 f32 matmul per tile.
  // A(16x4): col0=area_i, col1=1; B(4x16): row0=1, row1=area_j.
  const int wave = tid >> 5;
  const int lane = tid & 31;
  for (int t = wave; t < NTILE_C * NTILE_C; t += 8) {
    const int bi = t / NTILE_C;
    const int bj = t % NTILE_C;
    const int n = lane & 15;
    const int jj = bj * 16 + n;
#if defined(HAVE_WMMA_F32X4)
    v2f A, Bm;
    if (lane < 16) {               // lanes 0-15 hold K=0,1; lanes 16-31 K=2,3 (zero)
      A[0] = s_area[bi * 16 + lane]; A[1] = 1.0f;
      Bm[0] = 1.0f;                  Bm[1] = s_area[jj];
    } else {
      A[0] = 0.f; A[1] = 0.f; Bm[0] = 0.f; Bm[1] = 0.f;
    }
    v8f Cz = {0.f, 0.f, 0.f, 0.f, 0.f, 0.f, 0.f, 0.f};
    v8f U = __builtin_amdgcn_wmma_f32_16x16x4_f32(false, A, false, Bm, (short)0, Cz,
                                                  false, false);
#else
    v16h Ah, Bh;
    for (int q = 0; q < 16; ++q) { Ah[q] = (_Float16)0.f; Bh[q] = (_Float16)0.f; }
    if (lane < 16) {
      Ah[0] = (_Float16)s_area[bi * 16 + lane]; Ah[1] = (_Float16)1.0f;
      Bh[0] = (_Float16)1.0f;                   Bh[1] = (_Float16)s_area[jj];
    }
    v8f Cz = {0.f, 0.f, 0.f, 0.f, 0.f, 0.f, 0.f, 0.f};
    v8f U = __builtin_amdgcn_wmma_f32_16x16x32_f16(false, Ah, false, Bh, (short)0, Cz,
                                                   false, false);
#endif
    const float bx1 = s_box[jj*4+0], by1 = s_box[jj*4+1];
    const float bx2 = s_box[jj*4+2], by2 = s_box[jj*4+3];
    const int rowbase = bi * 16 + ((lane >= 16) ? 8 : 0);  // C/D VGPR layout
#pragma unroll
    for (int r = 0; r < 8; ++r) {
      const int ii = rowbase + r;
      const float ax1 = s_box[ii*4+0], ay1 = s_box[ii*4+1];
      const float ax2 = s_box[ii*4+2], ay2 = s_box[ii*4+3];
      float iw = fminf(ax2, bx2) - fmaxf(ax1, bx1); iw = fmaxf(iw, 0.f);
      float ih = fminf(ay2, by2) - fmaxf(ay1, by1); ih = fmaxf(ih, 0.f);
      const float inter = iw * ih;
      const float uni = U[r] - inter;         // area_i + area_j - inter
      const float iou = inter / fmaxf(uni, 1e-12f);
      if (iou > NMS_T)
        atomicOr(&s_sup[ii * 8 + (jj >> 5)], 1u << (jj & 31));
    }
  }
  __syncthreads();

  // Greedy sequential suppression on bitmasks (matches fori_loop semantics).
  if (tid == 0) {
    unsigned keep[7];
    for (int w = 0; w < 7; ++w) keep[w] = s_keep[w];
    for (int i = 0; i < TOPK_C; ++i) {
      if ((keep[i >> 5] >> (i & 31)) & 1u) {
        const int iw_ = i >> 5, ib = i & 31;
        for (int w = 0; w < 7; ++w) {
          unsigned g;                              // mask of j > i
          if (w < iw_) g = 0u;
          else if (w > iw_) g = 0xFFFFFFFFu;
          else g = (ib == 31) ? 0u : (0xFFFFFFFFu << (ib + 1));
          keep[w] &= ~(s_sup[i * 8 + w] & g);
        }
      }
    }
    int cnt = 0;
    for (int q = 0; q < TOPK_C; ++q) {
      if ((keep[q >> 5] >> (q & 31)) & 1u) s_pos[q] = cnt++;
      else s_pos[q] = -1;
    }
  }
  __syncthreads();

  // Compact kept rows: out[b, 1, pos] = (score, x1, y1, x2, y2).
  if (tid < TOPK_C) {
    const int p = s_pos[tid];
    if (p >= 0) {
      const size_t base = ((size_t)(b * 2 + 1) * TOPK_C + (size_t)p) * 5;
      out[base + 0] = s_score[tid];
      out[base + 1] = s_box[tid*4+0];
      out[base + 2] = s_box[tid*4+1];
      out[base + 3] = s_box[tid*4+2];
      out[base + 4] = s_box[tid*4+3];
    }
  }
}

// ---------------------------------------------------------------------------
extern "C" void kernel_launch(void* const* d_in, const int* in_sizes, int n_in,
                              void* d_out, int out_size, void* d_ws, size_t ws_size,
                              hipStream_t stream) {
  (void)in_sizes; (void)n_in; (void)ws_size;
  const float* loc   = (const float*)d_in[0];   // [64, 72192, 4]
  const float* conf  = (const float*)d_in[1];   // [64*72192, 2]
  const float* prior = (const float*)d_in[2];   // [72192, 4]
  float* out = (float*)d_out;                   // [64, 2, 200, 5]
  int*   wsI = (int*)d_ws;                      // 64*208 indices
  float* wsS = (float*)(wsI + BATCH_C * KPAD_C);// 64*208 scores

  zero_f32<<<(out_size + 255) / 256, 256, 0, stream>>>(out, out_size);
  topk_select<<<BATCH_C, 256, 0, stream>>>(conf, wsI, wsS, NUM_PRIORS_C);
  decode_nms<<<BATCH_C, 256, 0, stream>>>(loc, prior, wsI, wsS, out, NUM_PRIORS_C);
}
